// VNLinearLeakyReLU_Dual_76106820485879
// MI455X (gfx1250) — matrix operand order repaired and checked by reading, not compiled, not run on hardware
//
#include <hip/hip_runtime.h>

typedef _Float16 v8h  __attribute__((ext_vector_type(8)));
typedef _Float16 v16h __attribute__((ext_vector_type(16)));
typedef float    v8f  __attribute__((ext_vector_type(8)));

#define M_TOT   65536      // B*N*K
#define NBLK1   4096       // M_TOT / 16
#define OUTELEM 12582912   // B*C_OUT*3*N*K
#define VROW    200        // 192 + 8 half pad (LDS bank spread, 16B aligned)
#define WXROW   196

// ---------------------------------------------------------------- prep ------
__global__ void prep_weights(const float* __restrict__ Wf, const float* __restrict__ Wden,
                             const float* __restrict__ bden, const float* __restrict__ Wdird,
                             const float* __restrict__ Wdir, const float* __restrict__ proj,
                             const int* __restrict__ equiv,
                             _Float16* __restrict__ Wp, _Float16* __restrict__ Wd,
                             _Float16* __restrict__ Wdh, float* __restrict__ biasP) {
  int i = blockIdx.x * 256 + threadIdx.x;          // 0..36863
  int o3 = i / 192, j = i - o3 * 192;
  float ds = (equiv[0] != 0) ? 0.f : proj[0];
  float wden = Wden[i];
  float fe = ((j % 3) == (o3 % 3)) ? Wf[(o3 / 3) * 64 + (j / 3)] : 0.f;
  float fd = ((j % 3) == (o3 % 3)) ? Wdir[(o3 / 3) * 64 + (j / 3)] : 0.f;
  Wp[i]  = (_Float16)(ds * wden + fe);
  Wd[i]  = (_Float16)(ds * Wdird[i] + fd);
  Wdh[i] = (_Float16)wden;
  if (j == 0) biasP[o3] = ds * bden[o3];
}

// ------------------------------------------------------- wave 16x16 tile ----
__device__ __forceinline__ v8f wmma_tile(const _Float16* Abase, int rowbase,
                                         const _Float16* __restrict__ W, int col0, int lane) {
  v8f c = {0.f, 0.f, 0.f, 0.f, 0.f, 0.f, 0.f, 0.f};
  const int rl = lane & 15, hi = lane >> 4;
  const _Float16* rp = Abase + (size_t)(rowbase + rl) * VROW;
  const _Float16* wp = W + (size_t)(col0 + rl) * 192;
#pragma unroll
  for (int ks = 0; ks < 6; ++ks) {
    const int kb = ks * 32;
    // A 16x32 f16 layout: lane<16 halves = K[kb..kb+7],[kb+16..23]; lane>=16 +8
    v8h a0 = *(const v8h*)(rp + kb + hi * 8);
    v8h a1 = *(const v8h*)(rp + kb + 16 + hi * 8);
    // B 32x16 f16 layout: N = lane%16, K = [kb..kb+7],[kb+16..23]; lane>=16 +8
    v8h b0 = *(const v8h*)(wp + kb + hi * 8);
    v8h b1 = *(const v8h*)(wp + kb + 16 + hi * 8);
    v16h A = __builtin_shufflevector(a0, a1, 0,1,2,3,4,5,6,7,8,9,10,11,12,13,14,15);
    v16h B = __builtin_shufflevector(b0, b1, 0,1,2,3,4,5,6,7,8,9,10,11,12,13,14,15);
    c = __builtin_amdgcn_wmma_f32_16x16x32_f16(false, A, false, B, (short)0, c, false, false);
  }
  return c;
}

// ----------------------------------------------------------------- k1 -------
// One block = 16 locations. p -> d_out, d -> ws, Lie-bracket partials -> ws.
__global__ void __launch_bounds__(256)
k1_gemm(const float* __restrict__ x, const float* __restrict__ bden,
        const _Float16* __restrict__ Wp, const _Float16* __restrict__ Wd,
        const _Float16* __restrict__ Wdh, const float* __restrict__ biasP,
        float* __restrict__ pbuf, float* __restrict__ dbuf,
        float* __restrict__ ldpart, float* __restrict__ nrmpart) {
  __shared__ _Float16 Vt[16 * VROW];
  __shared__ _Float16 Tt[48 * VROW];
  __shared__ float WxS[16 * WXROW];
  __shared__ float red[256];

  const int tid = threadIdx.x, lane = tid & 31, wid = tid >> 5;
  const int m0 = blockIdx.x * 16;

  // stage V tile: v[j] = x[b, j/3, j%3, n, k]
  for (int e = tid; e < 16 * 192; e += 256) {
    int mloc = e / 192, j = e - mloc * 192;
    int m = m0 + mloc;
    int b = m >> 13, n = (m >> 4) & 511, k = m & 15;
    int cch = j / 3, a = j - 3 * cch;
    float v = x[((((size_t)b * 64 + cch) * 3 + a) * 512 + n) * 16 + k];
    Vt[mloc * VROW + j] = (_Float16)v;
  }
  __syncthreads();

  // T rows: signed gather of v via the 3x3 skew generators (GEN_IN)
  for (int e = tid; e < 48 * 192; e += 256) {
    int row = e / 192, q = e - row * 192;
    int mloc = row / 3, rr = row - 3 * mloc;
    const _Float16* vv = &Vt[mloc * VROW];
    _Float16 t = (_Float16)0.f;
    if (q < 64) {                    // gX: [1,2]=-1, [2,1]=+1
      if (rr == 1) t = -vv[3 * q + 2];
      else if (rr == 2) t = vv[3 * q + 1];
    } else if (q < 128) {            // gY: [0,2]=+1, [2,0]=-1
      int ch = q - 64;
      if (rr == 0) t = vv[3 * ch + 2];
      else if (rr == 2) t = -vv[3 * ch];
    } else {                         // gZ: [0,1]=-1, [1,0]=+1
      int ch = q - 128;
      if (rr == 0) t = -vv[3 * ch + 1];
      else if (rr == 1) t = vv[3 * ch];
    }
    Tt[row * VROW + q] = t;
  }
  __syncthreads();

  float accN = 0.f;
  // phase 1: 36 tiles = {p, d, Wx} x 12 column tiles
  for (int t = wid; t < 36; t += 8) {
    int role = t / 12, ct = t - role * 12, col0 = ct * 16;
    const _Float16* W = (role == 0) ? Wp : (role == 1) ? Wd : Wdh;
    v8f c = wmma_tile(Vt, 0, W, col0, lane);
    const int colN = col0 + (lane & 15), hi = lane >> 4;
    if (role == 2) {                 // Wx -> LDS (+ nrm partial)
      float bb = bden[colN];
#pragma unroll
      for (int r = 0; r < 8; ++r) {
        float val = c[r] + bb;
        WxS[(r + 8 * hi) * WXROW + colN] = val;
        accN += val * val;
      }
    } else {                         // p / d -> global [b,o,a,n,k]
      float bb = (role == 0) ? biasP[colN] : 0.f;
      float* out = (role == 0) ? pbuf : dbuf;
      int o = colN / 3, a = colN - 3 * o;
#pragma unroll
      for (int r = 0; r < 8; ++r) {
        int m = m0 + r + 8 * hi;
        int b = m >> 13, n = (m >> 4) & 511, k = m & 15;
        out[((((size_t)b * 64 + o) * 3 + a) * 512 + n) * 16 + k] = c[r] + bb;
      }
    }
  }
  __syncthreads();

  float accL = 0.f;
  // phase 2: WA tiles, fold (WA - AW)^2 immediately (AW = signed gather of Wx)
  for (int t = wid; t < 36; t += 8) {
    int mt = t / 12, ct = t - mt * 12, col0 = ct * 16;
    v8f c = wmma_tile(Tt, mt * 16, Wdh, col0, lane);
    const int q = col0 + (lane & 15), hi = lane >> 4;
    float bb = bden[q];
#pragma unroll
    for (int r = 0; r < 8; ++r) {
      int row = mt * 16 + r + 8 * hi;   // 0..47 = mloc*3 + rr
      int mloc = row / 3, rr = row - 3 * mloc;
      const float* wx = &WxS[mloc * WXROW];
      float aw = 0.f;
      if (q < 64) {
        if (rr == 1) aw = -wx[3 * q + 2];
        else if (rr == 2) aw = wx[3 * q + 1];
      } else if (q < 128) {
        int ch = q - 64;
        if (rr == 0) aw = wx[3 * ch + 2];
        else if (rr == 2) aw = -wx[3 * ch];
      } else {
        int ch = q - 128;
        if (rr == 0) aw = -wx[3 * ch + 1];
        else if (rr == 1) aw = wx[3 * ch];
      }
      float diff = c[r] + bb - aw;
      accL += diff * diff;
    }
  }
  __syncthreads();
  red[tid] = accL; __syncthreads();
  for (int s = 128; s > 0; s >>= 1) { if (tid < s) red[tid] += red[tid + s]; __syncthreads(); }
  if (tid == 0) ldpart[blockIdx.x] = red[0];
  __syncthreads();
  red[tid] = accN; __syncthreads();
  for (int s = 128; s > 0; s >>= 1) { if (tid < s) red[tid] += red[tid + s]; __syncthreads(); }
  if (tid == 0) nrmpart[blockIdx.x] = red[0];
}

// ----------------------------------------------------------------- k2 -------
__global__ void __launch_bounds__(256)
k2_bnstats(const float* __restrict__ pbuf, float* __restrict__ stats) {
  const int o = blockIdx.x, tid = threadIdx.x;
  __shared__ float r1[256], r2[256];
  float s = 0.f, s2 = 0.f;
  for (int idx = tid; idx < M_TOT; idx += 256) {
    int k = idx & 15, n = (idx >> 4) & 511, b = idx >> 13;
    size_t base = ((((size_t)b * 64 + o) * 3) * 512 + n) * 16 + k;
    float px = pbuf[base], py = pbuf[base + 8192], pz = pbuf[base + 16384];
    float nr = sqrtf(px * px + py * py + pz * pz) + 1e-6f;
    s += nr; s2 += nr * nr;
  }
  r1[tid] = s; r2[tid] = s2; __syncthreads();
  for (int st = 128; st > 0; st >>= 1) {
    if (tid < st) { r1[tid] += r1[tid + st]; r2[tid] += r2[tid + st]; }
    __syncthreads();
  }
  if (tid == 0) {
    float mean = r1[0] * (1.f / 65536.f);
    float var = r2[0] * (1.f / 65536.f) - mean * mean;
    stats[o] = mean;
    stats[64 + o] = rsqrtf(var + 1e-5f);
  }
}

// ----------------------------------------------------------------- k3 -------
__global__ void __launch_bounds__(256)
k3_apply(const float* __restrict__ dbuf, const float* __restrict__ stats,
         const float* __restrict__ gamma, const float* __restrict__ beta,
         float* __restrict__ outp) {
  int gid = blockIdx.x * 256 + threadIdx.x;     // (b,o,n,k)
  int k = gid & 15, n = (gid >> 4) & 511, o = (gid >> 13) & 63, b = gid >> 19;
  size_t base = ((((size_t)b * 64 + o) * 3) * 512 + n) * 16 + k;
  float p0 = outp[base], p1 = outp[base + 8192], p2 = outp[base + 16384];
  float d0 = dbuf[base], d1 = dbuf[base + 8192], d2 = dbuf[base + 16384];
  float nr = sqrtf(p0 * p0 + p1 * p1 + p2 * p2) + 1e-6f;
  float nbn = (nr - stats[o]) * stats[64 + o] * gamma[o] + beta[o];
  float sc = nbn / nr;
  p0 *= sc; p1 *= sc; p2 *= sc;
  float dot = p0 * d0 + p1 * d1 + p2 * d2;
  float o0 = p0, o1 = p1, o2 = p2;
  if (dot < 0.f) {
    float dsq = d0 * d0 + d1 * d1 + d2 * d2;
    float f = dot / (dsq + 1e-6f);
    o0 = 0.2f * p0 + 0.8f * (p0 - f * d0);
    o1 = 0.2f * p1 + 0.8f * (p1 - f * d1);
    o2 = 0.2f * p2 + 0.8f * (p2 - f * d2);
  }
  outp[base] = o0; outp[base + 8192] = o1; outp[base + 16384] = o2;
}

// ----------------------------------------------------------------- k4 -------
__global__ void __launch_bounds__(256)
k4_scalars(const float* __restrict__ ldpart, const float* __restrict__ nrmpart,
           const int* __restrict__ equiv, float* __restrict__ out) {
  __shared__ float r1[256], r2[256];
  int tid = threadIdx.x;
  float s1 = 0.f, s2 = 0.f;
  for (int i = tid; i < NBLK1; i += 256) { s1 += ldpart[i]; s2 += nrmpart[i]; }
  r1[tid] = s1; r2[tid] = s2; __syncthreads();
  for (int st = 128; st > 0; st >>= 1) {
    if (tid < st) { r1[tid] += r1[tid + st]; r2[tid] += r2[tid + st]; }
    __syncthreads();
  }
  if (tid == 0) {
    float sc = (equiv[0] != 0) ? 0.f : 2.f;   // source double-counts the lie term
    out[OUTELEM]     = sc * r1[0];
    out[OUTELEM + 1] = sc * r2[0];
  }
}

// -------------------------------------------------------------- launch ------
extern "C" void kernel_launch(void* const* d_in, const int* in_sizes, int n_in,
                              void* d_out, int out_size, void* d_ws, size_t ws_size,
                              hipStream_t stream) {
  const float* x      = (const float*)d_in[0];
  const float* W_feat = (const float*)d_in[1];
  const float* W_den  = (const float*)d_in[2];
  const float* b_den  = (const float*)d_in[3];
  const float* W_dird = (const float*)d_in[4];
  const float* W_dir  = (const float*)d_in[5];
  const float* gamma  = (const float*)d_in[6];
  const float* beta   = (const float*)d_in[7];
  const float* proj   = (const float*)d_in[8];
  const int*   equiv  = (const int*)d_in[9];
  float* out = (float*)d_out;

  char* w = (char*)d_ws;
  float*    dbuf   = (float*)w;                                   // 50,331,648 B
  _Float16* Wp     = (_Float16*)(w + (size_t)OUTELEM * 4);
  _Float16* Wd     = Wp + 36864;
  _Float16* Wdh    = Wd + 36864;
  float*    biasP  = (float*)(Wdh + 36864);
  float*    ldpart = biasP + 192;
  float*    nrmpart = ldpart + NBLK1;
  float*    stats  = nrmpart + NBLK1;

  prep_weights<<<144, 256, 0, stream>>>(W_feat, W_den, b_den, W_dird, W_dir,
                                        proj, equiv, Wp, Wd, Wdh, biasP);
  k1_gemm<<<NBLK1, 256, 0, stream>>>(x, b_den, Wp, Wd, Wdh, biasP,
                                     out, dbuf, ldpart, nrmpart);
  k2_bnstats<<<64, 256, 0, stream>>>(out, stats);
  k3_apply<<<16384, 256, 0, stream>>>(dbuf, stats, gamma, beta, out);
  k4_scalars<<<1, 256, 0, stream>>>(ldpart, nrmpart, equiv, out);
}
